// HierarchicalAttentionModule_43911745635108
// MI455X (gfx1250) — matrix operand order repaired and checked by reading, not compiled, not run on hardware
//
#include <hip/hip_runtime.h>

// ---------------------------------------------------------------------------
// HierarchicalAttentionModule for MI455X (gfx1250, wave32, WMMA bf16,
// async global->LDS staging when available)
// ---------------------------------------------------------------------------

#define CCH   256
#define HH    48
#define WW    48
#define NPOS  (HH*WW)      // 2304
#define NB    8

typedef __attribute__((ext_vector_type(16))) __bf16        v16bf;
typedef __attribute__((ext_vector_type(8)))  float         v8f;
typedef __attribute__((ext_vector_type(8)))  unsigned int  v8u;
// matches the builtin's parameter type reported by clang:
//   '__attribute__((__vector_size__(4 * sizeof(int)))) int *'
typedef __attribute__((__vector_size__(4 * sizeof(int)))) int vi4;

#if defined(__has_builtin)
#  if __has_builtin(__builtin_amdgcn_global_load_async_to_lds_b128)
#    define HAVE_ASYNC_LDS 1
#  else
#    define HAVE_ASYNC_LDS 0
#  endif
#else
#  define HAVE_ASYNC_LDS 0
#endif

// 16-byte global -> LDS copy (async on gfx1250 toolchains that expose it)
static __device__ __forceinline__ void cp16_g2l(const unsigned short* g,
                                                unsigned short* l) {
#if HAVE_ASYNC_LDS
  __builtin_amdgcn_global_load_async_to_lds_b128(
      (__attribute__((address_space(1))) vi4*)g,
      (__attribute__((address_space(3))) vi4*)l, 0, 0);
#else
  *(int4*)l = *(const int4*)g;
#endif
}

static __device__ __forceinline__ void wait_async0() {
#if HAVE_ASYNC_LDS
#  if __has_builtin(__builtin_amdgcn_s_wait_asynccnt)
  __builtin_amdgcn_s_wait_asynccnt(0);
#  else
  asm volatile("s_wait_asynccnt 0x0" ::: "memory");
#  endif
#endif
}

// fp32 -> bf16 (round-to-nearest-even)
static __device__ __forceinline__ unsigned short f2bf(float f) {
  unsigned int u = __builtin_bit_cast(unsigned int, f);
  u = (u + 0x7FFFu + ((u >> 16) & 1u)) >> 16;
  return (unsigned short)u;
}

// A fragment, 16x32 bf16, row-major [M][K] source with stride lda (ushorts).
// Lanes 0-15: K=0..15, lanes 16-31: K=16..31 (contiguous, matches B).
static __device__ __forceinline__ v16bf load_a_frag(const unsigned short* A,
                                                    int lda, int lane) {
  const int m = lane & 15, kh = lane >> 4;
  const unsigned short* r0 = A + m * lda + 16 * kh;
  const unsigned short* r1 = r0 + 8;
  v8u u;
#pragma unroll
  for (int v = 0; v < 4; ++v) u[v]     = *(const unsigned int*)(r0 + 2 * v);
#pragma unroll
  for (int v = 0; v < 4; ++v) u[4 + v] = *(const unsigned int*)(r1 + 2 * v);
  return __builtin_bit_cast(v16bf, u);
}

// B fragment, 32x16 bf16, from K-transposed source Bt[n][k] stride ldk.
// Contiguous k (as A): lanes 0-15 hold K=0..15, lanes 16-31 K=16..31, N = lane&15.
static __device__ __forceinline__ v16bf load_b_frag(const unsigned short* Bt,
                                                    int ldk, int lane) {
  const int n = lane & 15, kh = lane >> 4;
  const unsigned short* r = Bt + n * ldk + 16 * kh;
  v8u u;
#pragma unroll
  for (int v = 0; v < 8; ++v) u[v] = *(const unsigned int*)(r + 2 * v);
  return __builtin_bit_cast(v16bf, u);
}

static __device__ __forceinline__ v8f wmma_bf16(v16bf a, v16bf b, v8f c) {
  return __builtin_amdgcn_wmma_f32_16x16x32_bf16(false, a, false, b,
                                                 (short)0, c, false, false);
}

static __device__ __forceinline__ v8f vzero8() {
  v8f z;
#pragma unroll
  for (int i = 0; i < 8; ++i) z[i] = 0.f;
  return z;
}

// ---------------------------------------------------------------------------
// Kernel 0a: repack conv weights [O][C][taps] -> [tap][O][C] bf16,
//            and the 4 projection matrices -> [mat][O][C] bf16.
// ---------------------------------------------------------------------------
__global__ void prep_weights_kernel(const float* __restrict__ s1w,
                                    const float* __restrict__ s2w,
                                    const float* __restrict__ lw,
                                    const float* __restrict__ wq,
                                    const float* __restrict__ wk,
                                    const float* __restrict__ wv,
                                    const float* __restrict__ wa,
                                    unsigned short* __restrict__ w1T,
                                    unsigned short* __restrict__ w2T,
                                    unsigned short* __restrict__ wlT,
                                    unsigned short* __restrict__ wm4) {
  int idx = blockIdx.x * blockDim.x + threadIdx.x;
  const int n7 = 256 * 256 * 7, n9 = 256 * 256 * 9;
  if (idx < n7) {
    int t = idx / 65536, rem = idx & 65535, o = rem >> 8, c = rem & 255;
    w1T[idx] = f2bf(s1w[(o * 256 + c) * 7 + t]);
    w2T[idx] = f2bf(s2w[(o * 256 + c) * 7 + t]);
  }
  if (idx < n9) {
    int t = idx / 65536, rem = idx & 65535, o = rem >> 8, c = rem & 255;
    wlT[idx] = f2bf(lw[(o * 256 + c) * 9 + t]);
  }
  if (idx < 65536) {
    wm4[0 * 65536 + idx] = f2bf(wq[idx]);
    wm4[1 * 65536 + idx] = f2bf(wk[idx]);
    wm4[2 * 65536 + idx] = f2bf(wv[idx]);
    wm4[3 * 65536 + idx] = f2bf(wa[idx]);
  }
}

// ---------------------------------------------------------------------------
// Kernel 0b: x -> bf16 in both layouts: xbf [B][C][N], xTbf [B][N][C]
//            (LDS-tiled 32x32 transpose, coalesced both sides)
// ---------------------------------------------------------------------------
__global__ __launch_bounds__(256)
void xprep_kernel(const float* __restrict__ x,
                  unsigned short* __restrict__ xbf,
                  unsigned short* __restrict__ xTbf) {
  __shared__ unsigned short tile[32][33];
  const int pt = blockIdx.x;   // 72 position tiles
  const int ct = blockIdx.y;   // 8 channel tiles
  const int b  = blockIdx.z;
  const int tx = threadIdx.x & 31, ty = threadIdx.x >> 5;
  const size_t base = (size_t)b * CCH * NPOS;
#pragma unroll
  for (int i = 0; i < 32; i += 8) {
    int c = ct * 32 + ty + i, p = pt * 32 + tx;
    unsigned short v = f2bf(x[base + (size_t)c * NPOS + p]);
    xbf[base + (size_t)c * NPOS + p] = v;
    tile[ty + i][tx] = v;
  }
  __syncthreads();
#pragma unroll
  for (int i = 0; i < 32; i += 8) {
    int p = pt * 32 + ty + i, c = ct * 32 + tx;
    xTbf[(size_t)b * NPOS * CCH + (size_t)p * CCH + c] = tile[tx][ty + i];
  }
}

// ---------------------------------------------------------------------------
// Kernel 1: global (squeeze-excite) branch -> gatt[B][C]
// ---------------------------------------------------------------------------
__global__ __launch_bounds__(256)
void global_att_kernel(const float* __restrict__ x,
                       const float* __restrict__ g1w, const float* __restrict__ g1b,
                       const float* __restrict__ g2w, const float* __restrict__ g2b,
                       float* __restrict__ gatt) {
  const int b = blockIdx.x, tid = threadIdx.x;
  __shared__ float meanS[256];
  __shared__ float gS[32];
  const float* xb = x + (size_t)b * CCH * NPOS;
  float s = 0.f;
  for (int n = 0; n < NPOS; ++n) s += xb[(size_t)tid * NPOS + n];
  meanS[tid] = s * (1.0f / (float)NPOS);
  __syncthreads();
  if (tid < 32) {
    float a = g1b[tid];
    for (int c = 0; c < 256; ++c) a += meanS[c] * g1w[tid * 256 + c];
    gS[tid] = fmaxf(a, 0.f);
  }
  __syncthreads();
  float a = g2b[tid];
  for (int j = 0; j < 32; ++j) a += gS[j] * g2w[tid * 32 + j];
  gatt[b * 256 + tid] = 1.f / (1.f + __expf(-a));
}

// ---------------------------------------------------------------------------
// Kernel 2: fused Q/K/V/Anchor 1x1-conv GEMMs (bf16 WMMA)
//   q,k -> bf16 [B][N][C] ; v -> bf16 [B][C][N] ; anchor -> f32 [B][C][N]
// ---------------------------------------------------------------------------
__global__ __launch_bounds__(128)
void proj_kernel(const unsigned short* __restrict__ xTbf,
                 const unsigned short* __restrict__ wm4,
                 const float* __restrict__ bq, const float* __restrict__ bk,
                 const float* __restrict__ bv, const float* __restrict__ ba,
                 unsigned short* __restrict__ qT, unsigned short* __restrict__ kT,
                 unsigned short* __restrict__ vB, float* __restrict__ anchor) {
  const int nt = blockIdx.x;          // 36 position tiles
  const int my = blockIdx.y;          // mat(4) * mtile(4)
  const int b  = blockIdx.z;
  const int mat = my >> 2, mt = my & 3;
  const float* bias;
  switch (mat) {
    case 0:  bias = bq; break;
    case 1:  bias = bk; break;
    case 2:  bias = bv; break;
    default: bias = ba; break;
  }
  const int mbase = mt * 64, nbase = nt * 64;

  __shared__ __attribute__((aligned(16))) unsigned short As[64 * 72];
  __shared__ __attribute__((aligned(16))) unsigned short Bs[64 * 72];

  const int tid = threadIdx.x, lane = tid & 31, wave = tid >> 5;
  const int wm = (wave >> 1) * 32, wn = (wave & 1) * 32;

  v8f acc[2][2];
#pragma unroll
  for (int i = 0; i < 2; ++i)
#pragma unroll
    for (int j = 0; j < 2; ++j) acc[i][j] = vzero8();

  for (int kc = 0; kc < CCH; kc += 64) {
    __syncthreads();
    for (int i = tid; i < 64 * 8; i += 128) {          // weights [m][c], 16B units
      int m = i >> 3, ch = (i & 7) * 8;
      cp16_g2l(&wm4[((size_t)mat * 256 + mbase + m) * 256 + kc + ch],
               &As[m * 72 + ch]);
    }
    for (int i = tid; i < 64 * 8; i += 128) {          // x^T [n][c], 16B units
      int n = i >> 3, ch = (i & 7) * 8;
      cp16_g2l(&xTbf[((size_t)b * NPOS + nbase + n) * 256 + kc + ch],
               &Bs[n * 72 + ch]);
    }
    wait_async0();
    __syncthreads();
#pragma unroll
    for (int ks = 0; ks < 64; ks += 32) {
      v16bf a0 = load_a_frag(&As[(wm + 0)  * 72 + ks], 72, lane);
      v16bf a1 = load_a_frag(&As[(wm + 16) * 72 + ks], 72, lane);
      v16bf b0 = load_b_frag(&Bs[(wn + 0)  * 72 + ks], 72, lane);
      v16bf b1 = load_b_frag(&Bs[(wn + 16) * 72 + ks], 72, lane);
      acc[0][0] = wmma_bf16(a0, b0, acc[0][0]);
      acc[0][1] = wmma_bf16(a0, b1, acc[0][1]);
      acc[1][0] = wmma_bf16(a1, b0, acc[1][0]);
      acc[1][1] = wmma_bf16(a1, b1, acc[1][1]);
    }
  }
  const int n0 = lane & 15, rh = lane >> 4;
  const size_t bo = (size_t)b;
#pragma unroll
  for (int i = 0; i < 2; ++i)
#pragma unroll
    for (int j = 0; j < 2; ++j)
#pragma unroll
      for (int r = 0; r < 8; ++r) {
        int m = mbase + wm + i * 16 + r + 8 * rh;   // out channel
        int n = nbase + wn + j * 16 + n0;           // position
        float val = acc[i][j][r] + bias[m];
        if (mat == 0)      qT[(bo * NPOS + n) * CCH + m] = f2bf(val);
        else if (mat == 1) kT[(bo * NPOS + n) * CCH + m] = f2bf(val);
        else if (mat == 2) vB[(bo * CCH + m) * NPOS + n] = f2bf(val);
        else               anchor[(bo * CCH + m) * NPOS + n] = val;
      }
}

// ---------------------------------------------------------------------------
// Kernel 3: flash attention, head dim 256, seq 2304, per-batch.
//   One WG = 64 queries, 8 waves; K-blocks of 64 with online softmax.
//   Output attended f32 [B][C][N].
// ---------------------------------------------------------------------------
__global__ __launch_bounds__(256)
void attn_kernel(const unsigned short* __restrict__ qT,
                 const unsigned short* __restrict__ kT,
                 const unsigned short* __restrict__ vB,
                 float* __restrict__ attn_out) {
  const int qt = blockIdx.x;           // 36 query blocks
  const int b  = blockIdx.y;
  const int tid = threadIdx.x, lane = tid & 31, wave = tid >> 5;
  const int qbase = qt * 64;

  __shared__ __attribute__((aligned(16))) unsigned short Qs [64 * 264];
  __shared__ __attribute__((aligned(16))) unsigned short KsT[64 * 264];
  __shared__ __attribute__((aligned(16))) unsigned short Vs [256 * 72];
  __shared__ __attribute__((aligned(16))) float          Ss [64 * 66];
  __shared__ __attribute__((aligned(16))) unsigned short Ps [64 * 66];
  __shared__ float red1[64 * 5];
  __shared__ float red2[64 * 5];
  __shared__ float alphaS[64];
  __shared__ float lS[64];

  const unsigned short* qTb = qT + (size_t)b * NPOS * CCH;
  const unsigned short* kTb = kT + (size_t)b * NPOS * CCH;
  const unsigned short* vBb = vB + (size_t)b * CCH * NPOS;

  for (int i = tid; i < 64 * 32; i += 256) {           // Q block, resident
    int m = i >> 5, ch = (i & 31) * 8;
    cp16_g2l(&qTb[(size_t)(qbase + m) * CCH + ch], &Qs[m * 264 + ch]);
  }

  const int srow = tid & 63;                            // softmax row
  const int squad = tid >> 6;                           // quarter 0..3
  float m_i = -3.0e38f, l_i = 0.f;
  v8f acc[4][2];
#pragma unroll
  for (int i = 0; i < 4; ++i)
#pragma unroll
    for (int j = 0; j < 2; ++j) acc[i][j] = vzero8();

  for (int kb = 0; kb < NPOS; kb += 64) {
    __syncthreads();
    for (int i = tid; i < 64 * 32; i += 256) {         // K tile rows [col][c]
      int col = i >> 5, ch = (i & 31) * 8;
      cp16_g2l(&kTb[(size_t)(kb + col) * CCH + ch], &KsT[col * 264 + ch]);
    }
    for (int i = tid; i < 256 * 8; i += 256) {         // V tile [c][col]
      int c = i >> 3, ch = (i & 7) * 8;
      cp16_g2l(&vBb[(size_t)c * NPOS + kb + ch], &Vs[c * 72 + ch]);
    }
    wait_async0();
    __syncthreads();

    { // S = Q K^T : 16 tiles, 2 per wave
      const int qtl = wave >> 1;
      const int kt0 = (wave & 1) * 2;
      v8f s0 = vzero8(), s1 = vzero8();
#pragma unroll
      for (int c0 = 0; c0 < 256; c0 += 32) {
        v16bf a  = load_a_frag(&Qs[qtl * 16 * 264 + c0], 264, lane);
        v16bf b0 = load_b_frag(&KsT[(kt0 * 16)       * 264 + c0], 264, lane);
        v16bf b1 = load_b_frag(&KsT[((kt0 + 1) * 16) * 264 + c0], 264, lane);
        s0 = wmma_bf16(a, b0, s0);
        s1 = wmma_bf16(a, b1, s1);
      }
      const int n0 = lane & 15, rh = lane >> 4;
#pragma unroll
      for (int r = 0; r < 8; ++r) {
        int m = qtl * 16 + r + 8 * rh;
        Ss[m * 66 + kt0 * 16 + n0]       = s0[r];
        Ss[m * 66 + (kt0 + 1) * 16 + n0] = s1[r];
      }
    }
    __syncthreads();

    // ---- online softmax, 4 threads per row ----
    {
      const float* sp = &Ss[srow * 66 + squad * 16];
      float pmx = -3.0e38f;
#pragma unroll
      for (int j = 0; j < 16; ++j) pmx = fmaxf(pmx, sp[j]);
      red1[srow * 5 + squad] = pmx;
    }
    __syncthreads();
    float mnew;
    {
      const float* sp = &Ss[srow * 66 + squad * 16];
      float mx4 = fmaxf(fmaxf(red1[srow * 5 + 0], red1[srow * 5 + 1]),
                        fmaxf(red1[srow * 5 + 2], red1[srow * 5 + 3]));
      mnew = fmaxf(m_i, mx4);
      float psum = 0.f;
      unsigned short* pp = &Ps[srow * 66 + squad * 16];
#pragma unroll
      for (int j = 0; j < 16; ++j) {
        float p = __expf(sp[j] - mnew);
        psum += p;
        pp[j] = f2bf(p);
      }
      red2[srow * 5 + squad] = psum;
    }
    __syncthreads();
    {
      float s4 = (red2[srow * 5 + 0] + red2[srow * 5 + 1]) +
                 (red2[srow * 5 + 2] + red2[srow * 5 + 3]);
      float alpha = __expf(m_i - mnew);
      l_i = l_i * alpha + s4;
      m_i = mnew;
      if (squad == 0) alphaS[srow] = alpha;
    }
    __syncthreads();

    { // O = O*alpha + P V : wave owns 32 output channels
      const int cb = wave * 32;
      const int rh = lane >> 4;
#pragma unroll
      for (int qtl = 0; qtl < 4; ++qtl)
#pragma unroll
        for (int ct = 0; ct < 2; ++ct)
#pragma unroll
          for (int r = 0; r < 8; ++r)
            acc[qtl][ct][r] *= alphaS[qtl * 16 + r + 8 * rh];
#pragma unroll
      for (int ks = 0; ks < 64; ks += 32) {
        v16bf b0 = load_b_frag(&Vs[(cb + 0)  * 72 + ks], 72, lane);
        v16bf b1 = load_b_frag(&Vs[(cb + 16) * 72 + ks], 72, lane);
#pragma unroll
        for (int qtl = 0; qtl < 4; ++qtl) {
          v16bf a = load_a_frag(&Ps[qtl * 16 * 66 + ks], 66, lane);
          acc[qtl][0] = wmma_bf16(a, b0, acc[qtl][0]);
          acc[qtl][1] = wmma_bf16(a, b1, acc[qtl][1]);
        }
      }
    }
  }
  __syncthreads();
  if (tid < 64) lS[tid] = l_i;
  __syncthreads();

  float* ob = attn_out + (size_t)b * CCH * NPOS;
  const int n0 = lane & 15, rh = lane >> 4;
#pragma unroll
  for (int qtl = 0; qtl < 4; ++qtl)
#pragma unroll
    for (int ct = 0; ct < 2; ++ct)
#pragma unroll
      for (int r = 0; r < 8; ++r) {
        int ql = qtl * 16 + r + 8 * rh;
        int c  = wave * 32 + ct * 16 + n0;
        ob[(size_t)c * NPOS + qbase + ql] = acc[qtl][ct][r] / lS[ql];
      }
}

// ---------------------------------------------------------------------------
// Kernel 4: implicit-GEMM convolutions via tap-shifted WMMA GEMMs.
//   mode 0: (1,7) on xbf  -> relu    -> sb  bf16
//   mode 1: (7,1) on sb   -> sigmoid -> satt f32
//   mode 2: (3,3) on xbf  -> relu    -> localf f32
//   src is always bf16 [B][C][N].
// ---------------------------------------------------------------------------
__global__ __launch_bounds__(128)
void conv_kernel(const unsigned short* __restrict__ src,
                 const unsigned short* __restrict__ wT,   // [T][O][C] bf16
                 const float* __restrict__ bias,
                 void* __restrict__ dst,
                 int mode) {
  const int nt = blockIdx.x, mt = blockIdx.y, b = blockIdx.z;
  const int T = (mode == 2) ? 9 : 7;
  const int mbase = mt * 64, nbase = nt * 64;
  const int tid = threadIdx.x, lane = tid & 31, wave = tid >> 5;
  const int wm = (wave >> 1) * 32, wn = (wave & 1) * 32;

  __shared__ __attribute__((aligned(16))) unsigned short Wts[64 * 72];
  __shared__ __attribute__((aligned(16))) unsigned short Xs [64 * 72];

  v8f acc[2][2];
#pragma unroll
  for (int i = 0; i < 2; ++i)
#pragma unroll
    for (int j = 0; j < 2; ++j) acc[i][j] = vzero8();

  const size_t sbase = (size_t)b * CCH * NPOS;

  for (int tap = 0; tap < T; ++tap) {
    int dh, dw;
    if (mode == 0)      { dh = 0;           dw = tap - 3;     }
    else if (mode == 1) { dh = tap - 3;     dw = 0;           }
    else                { dh = tap / 3 - 1; dw = tap % 3 - 1; }

    for (int kc = 0; kc < CCH; kc += 64) {
      __syncthreads();
      for (int i = tid; i < 64 * 8; i += 128) {        // weights, 16B units
        int m = i >> 3, ch = (i & 7) * 8;
        cp16_g2l(&wT[((size_t)tap * 256 + mbase + m) * 256 + kc + ch],
                 &Wts[m * 72 + ch]);
      }
      for (int i = tid; i < 64 * 64; i += 128) {       // shifted x^T [n][c]
        int c = i >> 6, n = i & 63;
        int p = nbase + n;
        int h = p / WW + dh, w = p % WW + dw;
        unsigned short val = 0;
        if ((unsigned)h < (unsigned)HH && (unsigned)w < (unsigned)WW)
          val = src[sbase + (size_t)(kc + c) * NPOS + h * WW + w];
        Xs[n * 72 + c] = val;
      }
      wait_async0();
      __syncthreads();
#pragma unroll
      for (int ks = 0; ks < 64; ks += 32) {
        v16bf a0 = load_a_frag(&Wts[(wm + 0)  * 72 + ks], 72, lane);
        v16bf a1 = load_a_frag(&Wts[(wm + 16) * 72 + ks], 72, lane);
        v16bf b0 = load_b_frag(&Xs [(wn + 0)  * 72 + ks], 72, lane);
        v16bf b1 = load_b_frag(&Xs [(wn + 16) * 72 + ks], 72, lane);
        acc[0][0] = wmma_bf16(a0, b0, acc[0][0]);
        acc[0][1] = wmma_bf16(a0, b1, acc[0][1]);
        acc[1][0] = wmma_bf16(a1, b0, acc[1][0]);
        acc[1][1] = wmma_bf16(a1, b1, acc[1][1]);
      }
    }
  }
  const int n0 = lane & 15, rh = lane >> 4;
#pragma unroll
  for (int i = 0; i < 2; ++i)
#pragma unroll
    for (int j = 0; j < 2; ++j)
#pragma unroll
      for (int r = 0; r < 8; ++r) {
        int m = mbase + wm + i * 16 + r + 8 * rh;
        int n = nbase + wn + j * 16 + n0;
        float val = acc[i][j][r] + bias[m];
        size_t o = (size_t)b * CCH * NPOS + (size_t)m * NPOS + n;
        if (mode == 0)      ((unsigned short*)dst)[o] = f2bf(fmaxf(val, 0.f));
        else if (mode == 1) ((float*)dst)[o] = 1.f / (1.f + __expf(-val));
        else                ((float*)dst)[o] = fmaxf(val, 0.f);
      }
}

// ---------------------------------------------------------------------------
// Kernel 5: final combine
// ---------------------------------------------------------------------------
__global__ __launch_bounds__(256)
void combine_kernel(const float* __restrict__ x,    const float* __restrict__ gatt,
                    const float* __restrict__ satt, const float* __restrict__ localf,
                    const float* __restrict__ attn, const float* __restrict__ anchor,
                    float* __restrict__ out) {
  size_t idx = (size_t)blockIdx.x * blockDim.x + threadIdx.x;
  const size_t total = (size_t)NB * CCH * NPOS;
  if (idx >= total) return;
  int bc = (int)(idx / NPOS);                         // b*256 + c
  float xv = x[idx];
  out[idx] = xv * (1.f + gatt[bc] + satt[idx]) + localf[idx] + attn[idx] + anchor[idx];
}

// ---------------------------------------------------------------------------
extern "C" void kernel_launch(void* const* d_in, const int* in_sizes, int n_in,
                              void* d_out, int out_size, void* d_ws, size_t ws_size,
                              hipStream_t stream) {
  (void)in_sizes; (void)n_in; (void)out_size; (void)ws_size;
  const float* x   = (const float*)d_in[0];
  const float* wq  = (const float*)d_in[1];  const float* bq  = (const float*)d_in[2];
  const float* wk  = (const float*)d_in[3];  const float* bk  = (const float*)d_in[4];
  const float* wv  = (const float*)d_in[5];  const float* bv  = (const float*)d_in[6];
  const float* wa  = (const float*)d_in[7];  const float* ba  = (const float*)d_in[8];
  const float* g1w = (const float*)d_in[9];  const float* g1b = (const float*)d_in[10];
  const float* g2w = (const float*)d_in[11]; const float* g2b = (const float*)d_in[12];
  const float* s1w = (const float*)d_in[13]; const float* s1b = (const float*)d_in[14];
  const float* s2w = (const float*)d_in[15]; const float* s2b = (const float*)d_in[16];
  const float* lw  = (const float*)d_in[17]; const float* lb  = (const float*)d_in[18];
  float* out = (float*)d_out;

  char* ws = (char*)d_ws;
  size_t off = 0;
  auto walloc = [&](size_t bytes) -> void* {
    void* p = ws + off;
    off = (off + bytes + 255) & ~(size_t)255;
    return p;
  };
  const size_t ntot = (size_t)NB * CCH * NPOS;
  unsigned short* qT     = (unsigned short*)walloc(ntot * 2);
  unsigned short* kT     = (unsigned short*)walloc(ntot * 2);
  unsigned short* vB     = (unsigned short*)walloc(ntot * 2);
  unsigned short* xbf    = (unsigned short*)walloc(ntot * 2);
  unsigned short* xTbf   = (unsigned short*)walloc(ntot * 2);
  float*          anchor = (float*)walloc(ntot * 4);
  float*          attn   = (float*)walloc(ntot * 4);
  unsigned short* sb     = (unsigned short*)walloc(ntot * 2);
  float*          satt   = (float*)walloc(ntot * 4);
  float*          localf = (float*)walloc(ntot * 4);
  float*          gatt   = (float*)walloc((size_t)NB * 256 * 4);
  unsigned short* w1T    = (unsigned short*)walloc((size_t)7 * 65536 * 2);
  unsigned short* w2T    = (unsigned short*)walloc((size_t)7 * 65536 * 2);
  unsigned short* wlT    = (unsigned short*)walloc((size_t)9 * 65536 * 2);
  unsigned short* wm4    = (unsigned short*)walloc((size_t)4 * 65536 * 2);

  prep_weights_kernel<<<(256 * 256 * 9 + 255) / 256, 256, 0, stream>>>(
      s1w, s2w, lw, wq, wk, wv, wa, w1T, w2T, wlT, wm4);

  xprep_kernel<<<dim3(NPOS / 32, CCH / 32, NB), 256, 0, stream>>>(x, xbf, xTbf);

  global_att_kernel<<<NB, 256, 0, stream>>>(x, g1w, g1b, g2w, g2b, gatt);

  proj_kernel<<<dim3(NPOS / 64, 16, NB), 128, 0, stream>>>(
      xTbf, wm4, bq, bk, bv, ba, qT, kT, vB, anchor);

  attn_kernel<<<dim3(NPOS / 64, NB), 256, 0, stream>>>(qT, kT, vB, attn);

  conv_kernel<<<dim3(NPOS / 64, 4, NB), 128, 0, stream>>>(xbf, w1T, s1b, sb,     0);
  conv_kernel<<<dim3(NPOS / 64, 4, NB), 128, 0, stream>>>(sb,  w2T, s2b, satt,   1);
  conv_kernel<<<dim3(NPOS / 64, 4, NB), 128, 0, stream>>>(xbf, wlT, lb,  localf, 2);

  combine_kernel<<<(int)((ntot + 255) / 256), 256, 0, stream>>>(
      x, gatt, satt, localf, attn, anchor, out);
}
